// ResidualBlock_29996051596000
// MI455X (gfx1250) — matrix operand order repaired and checked by reading, not compiled, not run on hardware
//
#include <hip/hip_runtime.h>
#include <cstdint>
#include <cstddef>

#define B_ 16
#define C_ 256
#define L_ 2048
#define E_ 512
#define H_ 512
#define N_ 64

typedef __attribute__((ext_vector_type(16))) __bf16 v16bf;
typedef __attribute__((ext_vector_type(8)))  float  v8f;

union U4S8 { uint4 v; unsigned short s[8]; };

// ---------------- WMMA fragment helpers (A and B both use contiguous k = 16h + s, wave32) -------
// A-matrix 16x32 bf16, LDS tile As[M][K] row-major (lda = K+pad):
// lanes 0-15 -> M=lane, K=kb..kb+15 with kb = 16*(lane>>4).
__device__ __forceinline__ v16bf load_a_frag(const __bf16* As, int row0, int lane, int lda) {
  int m  = row0 + (lane & 15);
  int kb = (lane >> 4) * 16;
  const __bf16* p = As + m * lda + kb;
  v16bf a;
#pragma unroll
  for (int j = 0; j < 8; ++j) a[j] = p[j];
#pragma unroll
  for (int j = 0; j < 8; ++j) a[8 + j] = p[8 + j];
  return a;
}

// B-matrix 32x16 bf16 from an N-major LDS tile Bt[N][K] (ldb = K+pad):
// lane n holds K = kb..kb+15 (kb = 16*(lane>>4)) -> 16 contiguous halfs -> 2x b128.
__device__ __forceinline__ v16bf load_bt_frag(const __bf16* Bt, int col0, int lane, int ldb) {
  int n  = col0 + (lane & 15);
  int kb = (lane >> 4) * 16;
  const __bf16* p = Bt + n * ldb + kb;
  v16bf b;
#pragma unroll
  for (int j = 0; j < 16; ++j) b[j] = p[j];
  return b;
}

__device__ __forceinline__ v8f wmma_bf16(v16bf a, v16bf b, v8f c) {
  return __builtin_amdgcn_wmma_f32_16x16x32_bf16(false, a, false, b, (short)0, c, false, false);
}

// Stage a 32(K) x 128(N) bf16 tile from global (row-major, ld = L_) into LDS
// transposed as Bt[128][40].  Coalesced b128 global loads; packed b32 DS stores.
__device__ __forceinline__ void stage_bt(const __bf16* __restrict__ src, int k0, int n0,
                                         __bf16 (*Bt)[40], int tid) {
  int kp = (tid >> 4) * 2;      // 0,2,...,30
  int c  = (tid & 15) * 8;      // 0..120
  U4S8 lo, hi;
  lo.v = *(const uint4*)(src + (size_t)(k0 + kp) * L_ + n0 + c);
  hi.v = *(const uint4*)(src + (size_t)(k0 + kp + 1) * L_ + n0 + c);
#pragma unroll
  for (int i = 0; i < 8; ++i) {
    unsigned int pk = (unsigned int)lo.s[i] | ((unsigned int)hi.s[i] << 16);
    *(unsigned int*)&Bt[c + i][kp] = pk;
  }
}

// ---------------- small elementwise kernels --------------------------------
__global__ void cvt_bf16_kernel(const float* __restrict__ src, __bf16* __restrict__ dst, int n) {
  int i = blockIdx.x * blockDim.x + threadIdx.x;
  if (i < n) dst[i] = (__bf16)src[i];
}

// d[b,c] = emb[b,:] . dp_W[c,:] + dp_b[c]
__global__ void dproj_kernel(const float* __restrict__ emb, const float* __restrict__ W,
                             const float* __restrict__ bias, float* __restrict__ D) {
  int idx = blockIdx.x * blockDim.x + threadIdx.x;  // B_*C_ = 4096
  int b = idx >> 8, c = idx & 255;
  const float* e = emb + b * E_;
  const float* w = W + c * E_;
  float s = 0.f;
#pragma unroll 8
  for (int k = 0; k < E_; ++k) s += e[k] * w[k];
  D[idx] = s + bias[c];
}

// xd[b,c,l] = bf16(x[b,c,l] + d[b,c])
__global__ void add_diff_cvt_kernel(const float* __restrict__ x, const float* __restrict__ d,
                                    __bf16* __restrict__ xd) {
  int i = blockIdx.x * blockDim.x + threadIdx.x;    // B_*C_*L_
  xd[i] = (__bf16)(x[i] + d[i >> 11]);
}

// ---------------- GEMM 1: U = in_W[512x256] @ xd[b][256x2048] + in_b -------
__global__ void gemm_in_kernel(const __bf16* __restrict__ W, const __bf16* __restrict__ X,
                               const float* __restrict__ bias, float* __restrict__ U) {
  __shared__ __bf16 As[64][40];
  __shared__ __bf16 Bt[128][40];
  int b  = blockIdx.z;
  int m0 = blockIdx.y * 64;
  int n0 = blockIdx.x * 128;
  int tid = threadIdx.x, lane = tid & 31, wid = tid >> 5;
  int wm = wid & 3, wn = wid >> 2;
  const __bf16* Xb = X + (size_t)b * C_ * L_;
  v8f acc[4] = {};
  for (int k0 = 0; k0 < C_; k0 += 32) {
    { int r = tid >> 2, c = (tid & 3) * 8;
      *(uint4*)&As[r][c] = *(const uint4*)(W + (m0 + r) * C_ + k0 + c); }
    stage_bt(Xb, k0, n0, Bt, tid);
    if (k0 + 32 < C_) {
      __builtin_prefetch(W + (m0 + (tid >> 2)) * C_ + k0 + 32, 0, 0);
      __builtin_prefetch(Xb + (size_t)(k0 + 32 + (tid >> 4) * 2) * L_ + n0 + (tid & 15) * 8, 0, 0);
    }
    __syncthreads();
    v16bf a = load_a_frag(&As[0][0], wm * 16, lane, 40);
#pragma unroll
    for (int t = 0; t < 4; ++t) {
      v16bf bf = load_bt_frag(&Bt[0][0], wn * 64 + t * 16, lane, 40);
      acc[t] = wmma_bf16(a, bf, acc[t]);
    }
    __syncthreads();
  }
  int nlo = lane & 15, mhi = (lane >> 4) * 8;
#pragma unroll
  for (int t = 0; t < 4; ++t) {
    int col = n0 + wn * 64 + t * 16 + nlo;
#pragma unroll
    for (int r = 0; r < 8; ++r) {
      int row = m0 + wm * 16 + mhi + r;
      U[((size_t)b * H_ + row) * L_ + col] = acc[t][r] + bias[row];
    }
  }
}

// ---------------- GEMM 2: GLU(oW[1024x512] @ g[b][512x2048] + ob) ----------
__global__ void gemm_glu_kernel(const __bf16* __restrict__ W, const __bf16* __restrict__ G,
                                const float* __restrict__ bias, float* __restrict__ Y) {
  __shared__ __bf16 As[2][64][40];
  __shared__ __bf16 Bt[128][40];
  int b  = blockIdx.z;
  int m0 = blockIdx.y * 64;
  int n0 = blockIdx.x * 128;
  int tid = threadIdx.x, lane = tid & 31, wid = tid >> 5;
  int wm = wid & 3, wn = wid >> 2;
  const __bf16* Gb = G + (size_t)b * H_ * L_;
  v8f acc[2][4] = {};
  for (int k0 = 0; k0 < H_; k0 += 32) {
    { int r = tid >> 2, c = (tid & 3) * 8;
      *(uint4*)&As[0][r][c] = *(const uint4*)(W + (size_t)(m0 + r) * H_ + k0 + c);
      *(uint4*)&As[1][r][c] = *(const uint4*)(W + (size_t)(m0 + 512 + r) * H_ + k0 + c); }
    stage_bt(Gb, k0, n0, Bt, tid);
    if (k0 + 32 < H_) {
      __builtin_prefetch(W + (size_t)(m0 + (tid >> 2)) * H_ + k0 + 32, 0, 0);
      __builtin_prefetch(Gb + (size_t)(k0 + 32 + (tid >> 4) * 2) * L_ + n0 + (tid & 15) * 8, 0, 0);
    }
    __syncthreads();
    v16bf a0 = load_a_frag(&As[0][0][0], wm * 16, lane, 40);
    v16bf a1 = load_a_frag(&As[1][0][0], wm * 16, lane, 40);
#pragma unroll
    for (int t = 0; t < 4; ++t) {
      v16bf bf = load_bt_frag(&Bt[0][0], wn * 64 + t * 16, lane, 40);
      acc[0][t] = wmma_bf16(a0, bf, acc[0][t]);
      acc[1][t] = wmma_bf16(a1, bf, acc[1][t]);
    }
    __syncthreads();
  }
  int nlo = lane & 15, mhi = (lane >> 4) * 8;
#pragma unroll
  for (int t = 0; t < 4; ++t) {
    int col = n0 + wn * 64 + t * 16 + nlo;
#pragma unroll
    for (int r = 0; r < 8; ++r) {
      int row = m0 + wm * 16 + mhi + r;
      float av = acc[0][t][r] + bias[row];
      float bv = acc[1][t][r] + bias[row + 512];
      Y[((size_t)b * H_ + row) * L_ + col] = av * (1.f / (1.f + __expf(-bv)));
    }
  }
}

// ---------------- GEMM 3: out_W with fused residual/skip -------------------
__global__ void gemm_out_kernel(const __bf16* __restrict__ W, const __bf16* __restrict__ Hm,
                                const float* __restrict__ bias, const float* __restrict__ x,
                                float* __restrict__ out) {
  __shared__ __bf16 As[2][64][40];
  __shared__ __bf16 Bt[128][40];
  int b  = blockIdx.z;
  int m0 = blockIdx.y * 64;
  int n0 = blockIdx.x * 128;
  int tid = threadIdx.x, lane = tid & 31, wid = tid >> 5;
  int wm = wid & 3, wn = wid >> 2;
  const __bf16* Hb = Hm + (size_t)b * C_ * L_;
  v8f acc[2][4] = {};
  for (int k0 = 0; k0 < C_; k0 += 32) {
    { int r = tid >> 2, c = (tid & 3) * 8;
      *(uint4*)&As[0][r][c] = *(const uint4*)(W + (size_t)(m0 + r) * C_ + k0 + c);
      *(uint4*)&As[1][r][c] = *(const uint4*)(W + (size_t)(m0 + 256 + r) * C_ + k0 + c); }
    stage_bt(Hb, k0, n0, Bt, tid);
    if (k0 + 32 < C_) {
      __builtin_prefetch(W + (size_t)(m0 + (tid >> 2)) * C_ + k0 + 32, 0, 0);
      __builtin_prefetch(Hb + (size_t)(k0 + 32 + (tid >> 4) * 2) * L_ + n0 + (tid & 15) * 8, 0, 0);
    }
    __syncthreads();
    v16bf a0 = load_a_frag(&As[0][0][0], wm * 16, lane, 40);
    v16bf a1 = load_a_frag(&As[1][0][0], wm * 16, lane, 40);
#pragma unroll
    for (int t = 0; t < 4; ++t) {
      v16bf bf = load_bt_frag(&Bt[0][0], wn * 64 + t * 16, lane, 40);
      acc[0][t] = wmma_bf16(a0, bf, acc[0][t]);
      acc[1][t] = wmma_bf16(a1, bf, acc[1][t]);
    }
    __syncthreads();
  }
  const size_t BCL = (size_t)B_ * C_ * L_;
  int nlo = lane & 15, mhi = (lane >> 4) * 8;
#pragma unroll
  for (int t = 0; t < 4; ++t) {
    int col = n0 + wn * 64 + t * 16 + nlo;
#pragma unroll
    for (int r = 0; r < 8; ++r) {
      int row = m0 + wm * 16 + mhi + r;
      size_t idx = ((size_t)b * C_ + row) * L_ + col;
      float rv = acc[0][t][r] + bias[row];
      float sv = acc[1][t][r] + bias[row + 256];
      out[idx]       = (x[idx] + rv) * 0.70710678118654752f;
      out[BCL + idx] = sv;
    }
  }
}

// ---------------- S4 diagonal SSM via recurrence + GELU --------------------
// one wave32 per (b,h): lane holds states n=lane and n=lane+32.
__global__ void ssm_gelu_kernel(const float* __restrict__ U,
                                const float* __restrict__ log_dt,
                                const float* __restrict__ A_re, const float* __restrict__ A_im,
                                const float* __restrict__ C_re, const float* __restrict__ C_im,
                                const float* __restrict__ Dskip,
                                __bf16* __restrict__ G) {
  int wid  = threadIdx.x >> 5;
  int lane = threadIdx.x & 31;
  int gw = blockIdx.x * 8 + wid;     // b*H + h
  int b = gw >> 9;
  int h = gw & 511;
  float dt = __expf(log_dt[h]);
  float lre[2], lim[2], ctr[2], cti[2], sre[2] = {0.f, 0.f}, sim[2] = {0.f, 0.f};
#pragma unroll
  for (int q = 0; q < 2; ++q) {
    int n = lane + 32 * q;
    float ar = A_re[h * N_ + n], ai = A_im[h * N_ + n];
    float e  = __expf(dt * ar);
    float er = e * __cosf(dt * ai), ei = e * __sinf(dt * ai);
    lre[q] = er; lim[q] = ei;
    float zr = er - 1.f, zi = ei;
    float inv = 1.f / (ar * ar + ai * ai);
    float wr = (zr * ar + zi * ai) * inv, wi = (zi * ar - zr * ai) * inv;
    float cr = C_re[h * N_ + n], ci = C_im[h * N_ + n];
    ctr[q] = cr * wr - ci * wi;
    cti[q] = cr * wi + ci * wr;
  }
  float Dh = Dskip[h];
  const float* up = U + ((size_t)b * H_ + h) * L_;
  __bf16* gp = G + ((size_t)b * H_ + h) * L_;
  for (int l0 = 0; l0 < L_; l0 += 32) {
    float uch = up[l0 + lane];       // coalesced chunk, broadcast via shuffle
    float keep = 0.f;
    for (int s = 0; s < 32; ++s) {
      float ul = __shfl(uch, s, 32);
      float part = 0.f;
#pragma unroll
      for (int q = 0; q < 2; ++q) {
        float nr = lre[q] * sre[q] - lim[q] * sim[q] + ul;
        float ni = lre[q] * sim[q] + lim[q] * sre[q];
        sre[q] = nr; sim[q] = ni;
        part += ctr[q] * nr - cti[q] * ni;
      }
#pragma unroll
      for (int off = 16; off > 0; off >>= 1) part += __shfl_xor(part, off, 32);
      float y = 2.f * part + Dh * ul;
      float t = 0.7978845608f * (y + 0.044715f * y * y * y);
      float gl = 0.5f * y * (1.f + tanhf(t));
      keep = (s == lane) ? gl : keep;
    }
    gp[l0 + lane] = (__bf16)keep;   // coalesced store every 32 steps
  }
}

// ---------------- LayerNorm over channel dim -------------------------------
__global__ void ln_kernel(const float* __restrict__ Y, const float* __restrict__ g,
                          const float* __restrict__ bt, float* __restrict__ Z) {
  int idx = blockIdx.x * blockDim.x + threadIdx.x;  // b*L + l
  int b = idx >> 11, l = idx & 2047;
  const float* yp = Y + (size_t)b * H_ * L_ + l;
  float s = 0.f, s2 = 0.f;
  for (int h = 0; h < H_; ++h) { float v = yp[(size_t)h * L_]; s += v; s2 += v * v; }
  float mu = s * (1.f / H_);
  float rs = rsqrtf(s2 * (1.f / H_) - mu * mu + 1e-5f);
  float* zp = Z + (size_t)b * H_ * L_ + l;
  for (int h = 0; h < H_; ++h)
    zp[(size_t)h * L_] = (yp[(size_t)h * L_] - mu) * rs * g[h] + bt[h];
}

// LN2 fused with sigmoid(gate)*tanh(filt) -> bf16 [B,C,L]
__global__ void ln_gate_kernel(const float* __restrict__ Y, const float* __restrict__ g,
                               const float* __restrict__ bt, __bf16* __restrict__ Hm) {
  int idx = blockIdx.x * blockDim.x + threadIdx.x;
  int b = idx >> 11, l = idx & 2047;
  const float* yp = Y + (size_t)b * H_ * L_ + l;
  float s = 0.f, s2 = 0.f;
  for (int h = 0; h < H_; ++h) { float v = yp[(size_t)h * L_]; s += v; s2 += v * v; }
  float mu = s * (1.f / H_);
  float rs = rsqrtf(s2 * (1.f / H_) - mu * mu + 1e-5f);
  __bf16* hp = Hm + (size_t)b * C_ * L_ + l;
  for (int c = 0; c < C_; ++c) {
    float zg = (yp[(size_t)c * L_] - mu) * rs * g[c] + bt[c];
    float zf = (yp[(size_t)(c + 256) * L_] - mu) * rs * g[c + 256] + bt[c + 256];
    float sg = 1.f / (1.f + __expf(-zg));
    hp[(size_t)c * L_] = (__bf16)(sg * tanhf(zf));
  }
}

// ---------------- host launch ----------------------------------------------
extern "C" void kernel_launch(void* const* d_in, const int* in_sizes, int n_in,
                              void* d_out, int out_size, void* d_ws, size_t ws_size,
                              hipStream_t stream) {
  (void)in_sizes; (void)n_in; (void)out_size; (void)ws_size;
  const float* x      = (const float*)d_in[0];
  const float* emb    = (const float*)d_in[1];
  const float* dp_W   = (const float*)d_in[2];
  const float* dp_b   = (const float*)d_in[3];
  const float* in_W   = (const float*)d_in[4];
  const float* in_b   = (const float*)d_in[5];
  const float* out_W  = (const float*)d_in[6];
  const float* out_b  = (const float*)d_in[7];
  const float* ln1_g  = (const float*)d_in[8];
  const float* ln1_b  = (const float*)d_in[9];
  const float* ln2_g  = (const float*)d_in[10];
  const float* ln2_b  = (const float*)d_in[11];
  const float* s1_logdt = (const float*)d_in[12];
  const float* s1_Are   = (const float*)d_in[13];
  const float* s1_Aim   = (const float*)d_in[14];
  const float* s1_Cre   = (const float*)d_in[15];
  const float* s1_Cim   = (const float*)d_in[16];
  const float* s1_D     = (const float*)d_in[17];
  const float* s1_oW    = (const float*)d_in[18];
  const float* s1_ob    = (const float*)d_in[19];
  const float* s2_logdt = (const float*)d_in[20];
  const float* s2_Are   = (const float*)d_in[21];
  const float* s2_Aim   = (const float*)d_in[22];
  const float* s2_Cre   = (const float*)d_in[23];
  const float* s2_Cim   = (const float*)d_in[24];
  const float* s2_D     = (const float*)d_in[25];
  const float* s2_oW    = (const float*)d_in[26];
  const float* s2_ob    = (const float*)d_in[27];

  char* p = (char*)d_ws;
  auto carve = [&](size_t bytes) -> char* {
    char* r = p;
    p += (bytes + 255) & ~(size_t)255;
    return r;
  };
  float*  dproj = (float*)carve((size_t)B_ * C_ * 4);
  __bf16* wIn   = (__bf16*)carve((size_t)H_ * C_ * 2);
  __bf16* wOut  = (__bf16*)carve((size_t)2 * C_ * C_ * 2);
  __bf16* wO1   = (__bf16*)carve((size_t)2 * H_ * H_ * 2);
  __bf16* wO2   = (__bf16*)carve((size_t)2 * H_ * H_ * 2);
  __bf16* xdh   = (__bf16*)carve((size_t)B_ * C_ * L_ * 2);  // xd, later gated h
  __bf16* gbuf  = (__bf16*)carve((size_t)B_ * H_ * L_ * 2);
  float*  Ubuf  = (float*)carve((size_t)B_ * H_ * L_ * 4);
  float*  Zbuf  = (float*)carve((size_t)B_ * H_ * L_ * 4);

  // weights -> bf16
  cvt_bf16_kernel<<<(H_ * C_ + 255) / 256, 256, 0, stream>>>(in_W, wIn, H_ * C_);
  cvt_bf16_kernel<<<(2 * C_ * C_ + 255) / 256, 256, 0, stream>>>(out_W, wOut, 2 * C_ * C_);
  cvt_bf16_kernel<<<(2 * H_ * H_ + 255) / 256, 256, 0, stream>>>(s1_oW, wO1, 2 * H_ * H_);
  cvt_bf16_kernel<<<(2 * H_ * H_ + 255) / 256, 256, 0, stream>>>(s2_oW, wO2, 2 * H_ * H_);

  // diffusion projection + fuse into x, convert to bf16
  dproj_kernel<<<(B_ * C_) / 256, 256, 0, stream>>>(emb, dp_W, dp_b, dproj);
  add_diff_cvt_kernel<<<(B_ * C_ * L_) / 256, 256, 0, stream>>>(x, dproj, xdh);

  // in-proj GEMM
  gemm_in_kernel<<<dim3(L_ / 128, H_ / 64, B_), 256, 0, stream>>>(wIn, xdh, in_b, Ubuf);

  // S4 layer 1
  ssm_gelu_kernel<<<(B_ * H_) / 8, 256, 0, stream>>>(Ubuf, s1_logdt, s1_Are, s1_Aim,
                                                     s1_Cre, s1_Cim, s1_D, gbuf);
  gemm_glu_kernel<<<dim3(L_ / 128, H_ / 64, B_), 256, 0, stream>>>(wO1, gbuf, s1_ob, Ubuf);
  ln_kernel<<<(B_ * L_) / 256, 256, 0, stream>>>(Ubuf, ln1_g, ln1_b, Zbuf);

  // S4 layer 2
  ssm_gelu_kernel<<<(B_ * H_) / 8, 256, 0, stream>>>(Zbuf, s2_logdt, s2_Are, s2_Aim,
                                                     s2_Cre, s2_Cim, s2_D, gbuf);
  gemm_glu_kernel<<<dim3(L_ / 128, H_ / 64, B_), 256, 0, stream>>>(wO2, gbuf, s2_ob, Ubuf);
  ln_gate_kernel<<<(B_ * L_) / 256, 256, 0, stream>>>(Ubuf, ln2_g, ln2_b, xdh);

  // out-proj GEMM with fused residual/skip
  gemm_out_kernel<<<dim3(L_ / 128, C_ / 64, B_), 256, 0, stream>>>(wOut, xdh, out_b, x,
                                                                   (float*)d_out);
}
